// DEAttention_Module_68289980006554
// MI455X (gfx1250) — matrix operand order repaired and checked by reading, not compiled, not run on hardware
//
#include <hip/hip_runtime.h>
#include <hip/hip_bf16.h>
#include <stdint.h>

// MI455X / gfx1250: wave32, WMMA 16x16x32 bf16 -> f32 accumulate.
typedef __attribute__((ext_vector_type(16))) __bf16 v16bf;
typedef __attribute__((ext_vector_type(8)))  float  v8f;
typedef int v4i_ __attribute__((vector_size(16)));   // matches async-LDS builtin param

#define CD 512     // channels
#define SD 4096    // spatial (W*D) == attention N
#define NB 8       // batch
#define KSPLIT 4   // split-K for the gram (energy) GEMM
#define TILE_M 256
#define TILE_N 128
#define TILE_K 32
#define LDK 40     // LDS row stride in bf16 elements (32 K + 8 pad = 80B rows)

// ---- CDNA5 async global->LDS copies (ASYNCcnt path), guarded for portability ----
#if __has_builtin(__builtin_amdgcn_global_load_async_to_lds_b128)
#define HAS_ASYNC 1
#define ASYNC_COPY_B128(gp, lp)                                         \
  __builtin_amdgcn_global_load_async_to_lds_b128(                       \
      (__attribute__((address_space(1))) v4i_*)(gp),                    \
      (__attribute__((address_space(3))) v4i_*)(lp), 0, 0)
#if __has_builtin(__builtin_amdgcn_s_wait_asynccnt)
#define ASYNC_WAIT() __builtin_amdgcn_s_wait_asynccnt(0)
#else
#define ASYNC_WAIT() asm volatile("s_wait_asynccnt 0" ::: "memory")
#endif
#else
#define HAS_ASYNC 0
#define ASYNC_WAIT()
#endif

// ---------------- WMMA fragment loads from LDS ----------------
// A-matrix 16x32 bf16 (ISA 7.12.2): lane r=lane&15 is row M; half h=lane>>4:
//   elements 0..7 <-> K = 8h..8h+7 ; elements 8..15 <-> K = 16+8h..23+8h
__device__ __forceinline__ v16bf frag_a(const __bf16* As, int row, int h) {
  union { v16bf v; uint4 u[2]; } f;
  const __bf16* p = As + row * LDK;
  f.u[0] = *(const uint4*)(p + 8 * h);
  f.u[1] = *(const uint4*)(p + 16 + 8 * h);
  return f.v;
}
// B-matrix 32x16 bf16: lane r is column N; half h: elements 0..7 <-> K 8h..8h+7, 8..15 <-> K 16+8h..23+8h.
// Bs is stored n-major with K contiguous (B^T), so this is 2 x ds_load_b128.
__device__ __forceinline__ v16bf frag_b(const __bf16* Bs, int col, int h) {
  union { v16bf v; uint4 u[2]; } f;
  const __bf16* p = Bs + col * LDK;
  f.u[0] = *(const uint4*)(p + 8 * h);
  f.u[1] = *(const uint4*)(p + 16 + 8 * h);
  return f.v;
}

// One K-step per wave: 4 M-subtiles x 4 N-subtiles = 16 WMMAs (A and B reused 4x).
__device__ __forceinline__ void mma_step(const __bf16* As, const __bf16* Bs,
                                         int wm, int wn, int lane, v8f acc[4][4]) {
  const int r = lane & 15, h = lane >> 4;
  v16bf a[4], b[4];
#pragma unroll
  for (int mt = 0; mt < 4; ++mt) a[mt] = frag_a(As, wm * 64 + mt * 16 + r, h);
#pragma unroll
  for (int nt = 0; nt < 4; ++nt) b[nt] = frag_b(Bs, wn * 64 + nt * 16 + r, h);
#pragma unroll
  for (int mt = 0; mt < 4; ++mt)
#pragma unroll
    for (int nt = 0; nt < 4; ++nt)
      acc[mt][nt] = __builtin_amdgcn_wmma_f32_16x16x32_bf16(
          false, a[mt], false, b[nt], (short)0, acc[mt][nt], false, false);
}

#define ACC_INIT(acc)                    \
  v8f _z = {};                           \
  _Pragma("unroll")                      \
  for (int mt = 0; mt < 4; ++mt)         \
    _Pragma("unroll")                    \
    for (int nt = 0; nt < 4; ++nt) acc[mt][nt] = _z;

// ---------------- Kernel 1: Q/K/V projections ----------------
// Out_t[b,o,s] = sum_c W_t[o,c]*x[b,c,s] + bias_t[o], stored bf16 flat (o,s)
// (which *is* the (n,c) "q-prime" view used downstream).
__global__ __launch_bounds__(256)
void proj_kernel(const float* __restrict__ x,
                 const float* __restrict__ Wq, const float* __restrict__ bq,
                 const float* __restrict__ Wk, const float* __restrict__ bk,
                 const float* __restrict__ Wv, const float* __restrict__ bv,
                 __bf16* __restrict__ Q, __bf16* __restrict__ Kb_,
                 __bf16* __restrict__ V) {
  __shared__ __align__(16) __bf16 As[2][TILE_M * LDK];
  __shared__ __align__(16) __bf16 Bs[2][TILE_N * LDK];

  const int tid = threadIdx.x;
  const int s0 = blockIdx.x * TILE_N;
  const int o0 = blockIdx.y * TILE_M;
  const int z  = blockIdx.z;
  const int b  = z / 3;
  const int t  = z - 3 * b;

  const float* Wmat = (t == 0) ? Wq : (t == 1) ? Wk : Wv;
  const float* bias = (t == 0) ? bq : (t == 1) ? bk : bv;
  __bf16* Out = ((t == 0) ? Q : (t == 1) ? Kb_ : V) + (size_t)b * CD * SD;
  const float* Xb = x + (size_t)b * CD * SD;

  auto load_tiles = [&](int kk, int buf) {
    const int k0 = kk * TILE_K;
    // A: W[o0..+256, k0..+32] fp32 -> bf16, k-contiguous rows in LDS
#pragma unroll
    for (int i = 0; i < 8; ++i) {
      int idx = tid + i * 256;          // 0..2047
      int rr = idx >> 3, cq = idx & 7;  // 8 float4 per row
      float4 f = *(const float4*)(Wmat + (size_t)(o0 + rr) * CD + k0 + cq * 4);
      __bf16* d = &As[buf][rr * LDK + cq * 4];
      d[0] = (__bf16)f.x; d[1] = (__bf16)f.y; d[2] = (__bf16)f.z; d[3] = (__bf16)f.w;
    }
    // B: x[b, k0..+32 (c), s0..+128 (s)] coalesced fp32, transposed into LDS
#pragma unroll
    for (int i = 0; i < 4; ++i) {
      int idx = tid + i * 256;          // 0..1023
      int rr = idx >> 5, q = idx & 31;  // 32 float4 per row
      float4 f = *(const float4*)(Xb + (size_t)(k0 + rr) * SD + s0 + q * 4);
      Bs[buf][(q * 4 + 0) * LDK + rr] = (__bf16)f.x;
      Bs[buf][(q * 4 + 1) * LDK + rr] = (__bf16)f.y;
      Bs[buf][(q * 4 + 2) * LDK + rr] = (__bf16)f.z;
      Bs[buf][(q * 4 + 3) * LDK + rr] = (__bf16)f.w;
    }
  };

  const int wid = tid >> 5, lane = tid & 31;
  const int wm = wid & 3, wn = wid >> 2;  // 4x2 waves -> 256x128 tile

  v8f acc[4][4];
  ACC_INIT(acc)

  load_tiles(0, 0);
  __syncthreads();
  const int KT = CD / TILE_K;  // 16
  for (int kk = 0; kk < KT; ++kk) {
    const int cur = kk & 1;
    if (kk + 1 < KT) load_tiles(kk + 1, cur ^ 1);
    mma_step(As[cur], Bs[cur], wm, wn, lane, acc);
    __syncthreads();
  }

  const int r = lane & 15, h = lane >> 4;
#pragma unroll
  for (int mt = 0; mt < 4; ++mt)
#pragma unroll
    for (int nt = 0; nt < 4; ++nt) {
      const int col = s0 + wn * 64 + nt * 16 + r;
#pragma unroll
      for (int v = 0; v < 8; ++v) {
        const int row = o0 + wm * 64 + mt * 16 + 8 * h + v;
        Out[(size_t)row * SD + col] = (__bf16)(acc[mt][nt][v] + bias[row]);
      }
    }
}

// ---------------- Kernel 2: split-K energy partials ----------------
// part[b*4+p][i][j] = sum_{n in slice p} q'[n,i] * k'[n,j]
__global__ __launch_bounds__(256)
void energy_kernel(const __bf16* __restrict__ Q, const __bf16* __restrict__ Km,
                   float* __restrict__ part) {
  __shared__ __align__(16) __bf16 As[2][TILE_M * LDK];
  __shared__ __align__(16) __bf16 Bs[2][TILE_N * LDK];
  const int tid = threadIdx.x;
  const int j0 = blockIdx.x * TILE_N;
  const int i0 = blockIdx.y * TILE_M;
  const int z  = blockIdx.z;
  const int b  = z >> 2;
  const int p  = z & 3;
  const int nbase = p * (SD / KSPLIT);  // 1024-wide K slice
  const __bf16* Qb = Q  + (size_t)b * CD * SD;
  const __bf16* Kb = Km + (size_t)b * CD * SD;

  auto load_tiles = [&](int kk, int buf) {
    const int n0 = nbase + kk * TILE_K;  // K dim is n
    // As[i_local][n_local] = q'[n0+rr][i0+..]: coalesced b128, transposed LDS store
#pragma unroll
    for (int i = 0; i < 4; ++i) {
      int idx = tid + i * 256;          // 0..1023
      int rr = idx >> 5, q = idx & 31;  // 32 x (8 bf16) per row (256 i)
      uint4 u = *(const uint4*)(Qb + (size_t)(n0 + rr) * CD + i0 + q * 8);
      const __bf16* e = (const __bf16*)&u;
#pragma unroll
      for (int j = 0; j < 8; ++j) As[buf][(q * 8 + j) * LDK + rr] = e[j];
    }
#pragma unroll
    for (int i = 0; i < 2; ++i) {
      int idx = tid + i * 256;          // 0..511
      int rr = idx >> 4, q = idx & 15;  // 16 x (8 bf16) per row (128 j)
      uint4 u = *(const uint4*)(Kb + (size_t)(n0 + rr) * CD + j0 + q * 8);
      const __bf16* e = (const __bf16*)&u;
#pragma unroll
      for (int j = 0; j < 8; ++j) Bs[buf][(q * 8 + j) * LDK + rr] = e[j];
    }
  };

  const int wid = tid >> 5, lane = tid & 31;
  const int wm = wid & 3, wn = wid >> 2;
  v8f acc[4][4];
  ACC_INIT(acc)

  load_tiles(0, 0);
  __syncthreads();
  const int KT = (SD / KSPLIT) / TILE_K;  // 32
  for (int kk = 0; kk < KT; ++kk) {
    const int cur = kk & 1;
    if (kk + 1 < KT) load_tiles(kk + 1, cur ^ 1);
    mma_step(As[cur], Bs[cur], wm, wn, lane, acc);
    __syncthreads();
  }

  const int r = lane & 15, h = lane >> 4;
  float* Pb = part + (size_t)(b * KSPLIT + p) * CD * CD;
#pragma unroll
  for (int mt = 0; mt < 4; ++mt)
#pragma unroll
    for (int nt = 0; nt < 4; ++nt) {
      const int col = j0 + wn * 64 + nt * 16 + r;
#pragma unroll
      for (int v = 0; v < 8; ++v) {
        const int row = i0 + wm * 64 + mt * 16 + 8 * h + v;
        Pb[(size_t)row * CD + col] = acc[mt][nt][v];
      }
    }
}

// ---------------- Kernel 3: sum split-K partials + row softmax -> bf16 attn ----------------
__global__ __launch_bounds__(256)
void softmax_kernel(const float* __restrict__ part, __bf16* __restrict__ attn) {
  const int row = blockIdx.x;              // b*512 + i
  const int b = row >> 9, i = row & 511;
  const size_t sl = (size_t)CD * CD;
  const float* e = part + (size_t)(b * KSPLIT) * sl + (size_t)i * CD;
  __shared__ float red[256];
  const int t = threadIdx.x;
  float v0 = e[t]       + e[t + sl]       + e[t + 2 * sl]       + e[t + 3 * sl];
  float v1 = e[t + 256] + e[t + 256 + sl] + e[t + 256 + 2 * sl] + e[t + 256 + 3 * sl];
  red[t] = fmaxf(v0, v1);
  __syncthreads();
  for (int s = 128; s > 0; s >>= 1) {
    if (t < s) red[t] = fmaxf(red[t], red[t + s]);
    __syncthreads();
  }
  const float rmax = red[0];
  __syncthreads();
  float e0 = __expf(v0 - rmax), e1 = __expf(v1 - rmax);
  red[t] = e0 + e1;
  __syncthreads();
  for (int s = 128; s > 0; s >>= 1) {
    if (t < s) red[t] += red[t + s];
    __syncthreads();
  }
  const float inv = 1.0f / red[0];
  attn[(size_t)row * CD + t]       = (__bf16)(e0 * inv);
  attn[(size_t)row * CD + t + 256] = (__bf16)(e1 * inv);
}

// ---------------- Kernel 4: out'[n,i] = sum_j v'[n,j]*attn[i,j]; fused gamma*out'+x ----------------
__global__ __launch_bounds__(256)
void out_kernel(const __bf16* __restrict__ V, const __bf16* __restrict__ attn,
                const float* __restrict__ x, const float* __restrict__ gamma,
                float* __restrict__ out) {
  __shared__ __align__(16) __bf16 As[2][TILE_M * LDK];
  __shared__ __align__(16) __bf16 Bs[2][TILE_N * LDK];
  const int tid = threadIdx.x;
  const int i0 = blockIdx.x * TILE_N;  // output col = i
  const int n0 = blockIdx.y * TILE_M;  // output row = n
  const int b  = blockIdx.z;
  const __bf16* Vb = V    + (size_t)b * CD * SD;
  const __bf16* Ab = attn + (size_t)b * CD * CD;

  // Both operands are straight b128 copies: use CDNA5 async global->LDS.
  auto load_tiles = [&](int kk, int buf) {
    const int k0 = kk * TILE_K;  // K dim is j
#pragma unroll
    for (int i = 0; i < 4; ++i) {
      int idx = tid + i * 256;          // 0..1023
      int rr = idx >> 2, q = idx & 3;   // 4 x (8 bf16) per row (256 n)
      const __bf16* gp = Vb + (size_t)(n0 + rr) * CD + k0 + q * 8;
      __bf16* lp = &As[buf][rr * LDK + q * 8];
#if HAS_ASYNC
      ASYNC_COPY_B128(gp, lp);
#else
      *(uint4*)lp = *(const uint4*)gp;
#endif
    }
#pragma unroll
    for (int i = 0; i < 2; ++i) {
      int idx = tid + i * 256;          // 0..511
      int rr = idx >> 2, q = idx & 3;   // 4 x (8 bf16) per row (128 i)
      const __bf16* gp = Ab + (size_t)(i0 + rr) * CD + k0 + q * 8;
      __bf16* lp = &Bs[buf][rr * LDK + q * 8];
#if HAS_ASYNC
      ASYNC_COPY_B128(gp, lp);
#else
      *(uint4*)lp = *(const uint4*)gp;
#endif
    }
  };

  const int wid = tid >> 5, lane = tid & 31;
  const int wm = wid & 3, wn = wid >> 2;
  v8f acc[4][4];
  ACC_INIT(acc)

  load_tiles(0, 0);
  ASYNC_WAIT();
  __syncthreads();
  const int KT = CD / TILE_K;  // 16
  for (int kk = 0; kk < KT; ++kk) {
    const int cur = kk & 1;
    if (kk + 1 < KT) load_tiles(kk + 1, cur ^ 1);  // overlaps the WMMAs below
    mma_step(As[cur], Bs[cur], wm, wn, lane, acc);
    ASYNC_WAIT();
    __syncthreads();
  }

  const float g = gamma[0];
  const size_t base = (size_t)b * CD * SD;
  const int r = lane & 15, h = lane >> 4;
#pragma unroll
  for (int mt = 0; mt < 4; ++mt)
#pragma unroll
    for (int nt = 0; nt < 4; ++nt) {
      const int col = i0 + wn * 64 + nt * 16 + r;
#pragma unroll
      for (int v = 0; v < 8; ++v) {
        const int row = n0 + wm * 64 + mt * 16 + 8 * h + v;
        // out' flat (n*512+i) is exactly the (c,w,d) flat layout of the output
        const size_t tix = base + (size_t)row * CD + col;
        out[tix] = g * acc[mt][nt][v] + x[tix];
      }
    }
}

// ---------------- Launch ----------------
extern "C" void kernel_launch(void* const* d_in, const int* in_sizes, int n_in,
                              void* d_out, int out_size, void* d_ws, size_t ws_size,
                              hipStream_t stream) {
  const float* x     = (const float*)d_in[0];
  const float* Wq    = (const float*)d_in[1];
  const float* bq    = (const float*)d_in[2];
  const float* Wk    = (const float*)d_in[3];
  const float* bk    = (const float*)d_in[4];
  const float* Wv    = (const float*)d_in[5];
  const float* bv    = (const float*)d_in[6];
  const float* gamma = (const float*)d_in[7];
  float* out = (float*)d_out;

  char* ws = (char*)d_ws;
  const size_t qkvBytes  = (size_t)NB * CD * SD * sizeof(__bf16);           // 32 MB each
  const size_t partBytes = (size_t)NB * KSPLIT * CD * CD * sizeof(float);   // 32 MB
  __bf16* Q    = (__bf16*)(ws);
  __bf16* K    = (__bf16*)(ws + qkvBytes);
  __bf16* V    = (__bf16*)(ws + 2 * qkvBytes);
  float*  part = (float*)(ws + 3 * qkvBytes);
  __bf16* attn = (__bf16*)(ws + 3 * qkvBytes + partBytes);                  // 4 MB

  proj_kernel<<<dim3(SD / TILE_N, CD / TILE_M, NB * 3), 256, 0, stream>>>(
      x, Wq, bq, Wk, bk, Wv, bv, Q, K, V);
  energy_kernel<<<dim3(CD / TILE_N, CD / TILE_M, NB * KSPLIT), 256, 0, stream>>>(Q, K, part);
  softmax_kernel<<<dim3(NB * CD), 256, 0, stream>>>(part, attn);
  out_kernel<<<dim3(CD / TILE_N, SD / TILE_M, NB), 256, 0, stream>>>(V, attn, x, gamma, out);
}
